// CausalSelfAttention_34565896798621
// MI455X (gfx1250) — matrix operand order repaired and checked by reading, not compiled, not run on hardware
//
#include <hip/hip_runtime.h>
#include <hip/hip_bf16.h>
#include <stdint.h>

#define N_EMBD   1024
#define N_HEAD   16
#define HEAD_DIM 64
#define BATCH    4
#define SEQ      2048
#define MROWS    (BATCH * SEQ)   /* 8192 */

typedef __attribute__((ext_vector_type(16))) __bf16 v16bf;
typedef __attribute__((ext_vector_type(8)))  float  v8f;

union FragBF { v16bf v; uint4 q[2]; };
union H4     { uint2 u; __bf16 h[4]; };

#if __has_builtin(__builtin_amdgcn_global_load_async_to_lds_b128)
#define ASYNC_LDS 1
#else
#define ASYNC_LDS 0
#endif

// Pointer element type the async-LDS builtin expects: GCC-style int4 vector.
typedef int gv4i __attribute__((vector_size(16)));
typedef __attribute__((address_space(1))) gv4i gv4i_as1;  // global
typedef __attribute__((address_space(3))) gv4i gv4i_as3;  // LDS

// Copy 16B global -> LDS. Async path uses the CDNA5 ASYNCcnt DMA (no VGPR round trip).
static __device__ inline void cp16_to_lds(const __bf16* g, __bf16* l) {
#if ASYNC_LDS
  __builtin_amdgcn_global_load_async_to_lds_b128((gv4i_as1*)g, (gv4i_as3*)l, 0, 0);
#else
  *(uint4*)l = *(const uint4*)g;
#endif
}

template <int N>
static __device__ inline void wait_async() {
#if ASYNC_LDS
#if __has_builtin(__builtin_amdgcn_s_wait_asynccnt)
  __builtin_amdgcn_s_wait_asynccnt((unsigned short)N);
#else
  asm volatile("s_wait_asynccnt %0" ::"i"(N) : "memory");
#endif
#endif
}

static __device__ inline v8f wmma_bf16f32(v16bf a, v16bf b, v8f c) {
  return __builtin_amdgcn_wmma_f32_16x16x32_bf16(false, a, false, b, (short)0, c, false, false);
}

// 16-element bf16 fragment = two aligned 16B chunks from a row pointer.
static __device__ inline v16bf ldfrag(const __bf16* p, int off0, int off1) {
  FragBF f;
  f.q[0] = *(const uint4*)(p + off0);
  f.q[1] = *(const uint4*)(p + off1);
  return f.v;
}

/* ---------------- fp32 -> bf16 (row-major) ---------------- */
__global__ void __launch_bounds__(256)
cvt_f32_bf16_kernel(const float* __restrict__ src, __bf16* __restrict__ dst, int n) {
  int i = (blockIdx.x * blockDim.x + threadIdx.x) * 4;
  int stride = gridDim.x * blockDim.x * 4;
  for (; i + 3 < n; i += stride) {
    float4 f = *(const float4*)(src + i);
    H4 o;
    o.h[0] = (__bf16)f.x; o.h[1] = (__bf16)f.y;
    o.h[2] = (__bf16)f.z; o.h[3] = (__bf16)f.w;
    *(uint2*)(dst + i) = o.u;
  }
}

/* ---------------- fp32 [K][N] -> bf16 [N][K] (one-time weight transpose).
   K,N compile-time so the div/mod lower to shifts/magic-mul, not a runtime divide. -------- */
template <int K, int N>
__global__ void __launch_bounds__(256)
cvt_transpose_kernel(const float* __restrict__ src, __bf16* __restrict__ dst) {
  int i = blockIdx.x * blockDim.x + threadIdx.x;
  const int total = K * N;
  const int stride = gridDim.x * blockDim.x;
  for (; i < total; i += stride) {
    int k = i / N, n = i - k * N;
    dst[(size_t)n * K + k] = (__bf16)src[i];
  }
}

/* ---------------- tiled bf16 GEMM: C[8192, N] = A[8192,1024] * W[1024,N] + bias
   W supplied pre-transposed as WT[N][1024]. Block tile 128x256, BK=32,
   8 waves x (64x64), double-buffered LDS filled by async global->LDS DMA.
   MODE 0: scatter bf16 q/k into [B,H,T,64], v pre-transposed into [B,H,64,T]
   MODE 1: fp32 row-major output + bias                                        ---------------- */
template <int MODE>
__global__ void __launch_bounds__(256)
gemm_bf16_kernel(const __bf16* __restrict__ A, const __bf16* __restrict__ WT,
                 const float* __restrict__ bias, int N,
                 __bf16* __restrict__ qo, __bf16* __restrict__ ko, __bf16* __restrict__ vo,
                 float* __restrict__ fo) {
  const int K = N_EMBD;                 // 1024
  __shared__ __bf16 sA[2][128 * 40];    // [row][k], BK=32 padded to 40
  __shared__ __bf16 sBT[2][256 * 40];   // [n][k]

  const int tid   = threadIdx.x;
  const int lane  = tid & 31;
  const int wave  = tid >> 5;           // 0..7
  const int wm    = wave >> 2;          // 0..1  -> 64 rows each
  const int wn    = wave & 3;           // 0..3  -> 64 cols each
  const int bm    = blockIdx.y * 128;
  const int bn    = blockIdx.x * 256;
  const int r16   = lane & 15;
  const int half8 = (lane < 16) ? 0 : 8;
  const int a0    = (lane < 16) ? 0 : 8;   // A-frag K-chunk pattern
  const int b0    = (lane < 16) ? 0 : 16;  // unused: B-frag uses the a0 pattern like A

  v8f acc[4][4];
#pragma unroll
  for (int m = 0; m < 4; ++m)
#pragma unroll
    for (int n = 0; n < 4; ++n) acc[m][n] = (v8f)0.0f;

  auto stage = [&](int kt, int buf) {
#pragma unroll
    for (int i = 0; i < 2; ++i) {          // A tile: 128x32 = 512 chunks of 8 elems
      int c = tid + i * 256;
      int row = c >> 2, c4 = c & 3;
      cp16_to_lds(A + (size_t)(bm + row) * K + kt + c4 * 8, &sA[buf][row * 40 + c4 * 8]);
    }
#pragma unroll
    for (int i = 0; i < 4; ++i) {          // B tile: 256x32 = 1024 chunks
      int c = tid + i * 256;
      int nr = c >> 2, c4 = c & 3;
      cp16_to_lds(WT + (size_t)(bn + nr) * K + kt + c4 * 8, &sBT[buf][nr * 40 + c4 * 8]);
    }
  };

  const int NT = K / 32;                   // 32 iterations
  stage(0, 0);
  for (int it = 0; it < NT; ++it) {
    wait_async<0>();                       // own async copies for tile `it` retired
    __syncthreads();                       // everyone's copies visible; last readers of buf^1 done
    if (it + 1 < NT) stage((it + 1) * 32, (it + 1) & 1);  // overlaps with compute below

    const __bf16* bufA = sA[it & 1];
    const __bf16* bufB = sBT[it & 1];
    v16bf af[4], bfr[4];
#pragma unroll
    for (int m = 0; m < 4; ++m)
      af[m] = ldfrag(bufA + (wm * 64 + m * 16 + r16) * 40, a0, a0 + 16);
#pragma unroll
    for (int n = 0; n < 4; ++n)
      bfr[n] = ldfrag(bufB + (wn * 64 + n * 16 + r16) * 40, a0, a0 + 16);
#pragma unroll
    for (int m = 0; m < 4; ++m)
#pragma unroll
      for (int n = 0; n < 4; ++n)
        acc[m][n] = wmma_bf16f32(af[m], bfr[n], acc[m][n]);
  }

  // ---- epilogue ----
#pragma unroll
  for (int m = 0; m < 4; ++m) {
#pragma unroll
    for (int n = 0; n < 4; ++n) {
      const int col = bn + wn * 64 + n * 16 + r16;
      const float bv = bias[col];
#pragma unroll
      for (int e = 0; e < 8; ++e) {
        const int row = bm + wm * 64 + m * 16 + e + half8;
        const float val = acc[m][n][e] + bv;
        if (MODE == 0) {
          int sel = col >> 10, cc = col & 1023;
          int h = cc >> 6, d = cc & 63;
          int b = row >> 11, t = row & 2047;
          __bf16 hv = (__bf16)val;
          if (sel == 0)
            qo[((size_t)(b * N_HEAD + h) * SEQ + t) * HEAD_DIM + d] = hv;
          else if (sel == 1)
            ko[((size_t)(b * N_HEAD + h) * SEQ + t) * HEAD_DIM + d] = hv;
          else  // V stored transposed: [B,H,hd,T] so attention can DMA it straight to LDS
            vo[((size_t)(b * N_HEAD + h) * HEAD_DIM + d) * SEQ + t] = hv;
        } else {
          fo[(size_t)row * N + col] = val;
        }
      }
    }
  }
}

/* ---------------- flash attention: one block = (b*H + h, 64 q-rows), 4 waves x 16 rows ------- */
__global__ void __launch_bounds__(128)
attn_kernel(const __bf16* __restrict__ qb, const __bf16* __restrict__ kb,
            const __bf16* __restrict__ vb, __bf16* __restrict__ ob) {
  __shared__ __bf16 sK[2][32 * 72];    // [key][d] padded
  __shared__ __bf16 sVT[2][64 * 40];   // [d][key] (V arrives pre-transposed from global)
  __shared__ __bf16 sP[4][16 * 40];    // per-wave P bounce (C-layout -> A-layout)

  const int tid   = threadIdx.x;
  const int lane  = tid & 31;
  const int wave  = tid >> 5;          // 0..3
  const int bh    = blockIdx.x >> 5;   // b*16 + h
  const int qblk  = blockIdx.x & 31;
  const int qbase = qblk * 64;
  const int r16   = lane & 15;
  const int half8 = (lane < 16) ? 0 : 8;
  const int a0    = (lane < 16) ? 0 : 8;
  const int b0    = (lane < 16) ? 0 : 16;

  const __bf16* qrow = qb + ((size_t)bh * SEQ + qbase + wave * 16 + r16) * HEAD_DIM;
  v16bf qf[2];
  qf[0] = ldfrag(qrow, a0, a0 + 16);
  qf[1] = ldfrag(qrow, 32 + a0, 32 + a0 + 16);

  v8f o[4];
#pragma unroll
  for (int nf = 0; nf < 4; ++nf) o[nf] = (v8f)0.0f;
  float m_i[8], l_i[8];
#pragma unroll
  for (int e = 0; e < 8; ++e) { m_i[e] = -1e30f; l_i[e] = 0.0f; }

  auto stageKV = [&](int kbase, int buf) {
#pragma unroll
    for (int i = 0; i < 2; ++i) {
      int c = tid + i * 128;                 // 256 chunks each for K and VT tiles
      int row = c >> 3, c4 = c & 7;          // K tile: 32 rows x 8 chunks
      cp16_to_lds(kb + ((size_t)bh * SEQ + kbase + row) * HEAD_DIM + c4 * 8,
                  &sK[buf][row * 72 + c4 * 8]);
      int d = c >> 2, c42 = c & 3;           // VT tile: 64 rows x 4 chunks
      cp16_to_lds(vb + ((size_t)bh * HEAD_DIM + d) * SEQ + kbase + c42 * 8,
                  &sVT[buf][d * 40 + c42 * 8]);
    }
  };

  __bf16* sPw = &sP[wave][0];
  const float scale = 0.125f;                // 1/sqrt(64)
  const int nkt = qblk * 2 + 2;              // causal tile count (32 keys/tile)

  stageKV(0, 0);
  for (int kt = 0; kt < nkt; ++kt) {
    const int kbase = kt * 32;
    wait_async<0>();
    __syncthreads();
    if (kt + 1 < nkt) stageKV((kt + 1) * 32, (kt + 1) & 1);  // overlaps with compute

    const __bf16* bK = sK[kt & 1];
    const __bf16* bV = sVT[kt & 1];

    // ---- S = Q K^T ----
    v8f s0 = (v8f)0.0f, s1 = (v8f)0.0f;
#pragma unroll
    for (int ks = 0; ks < 2; ++ks) {
      v16bf bk0 = ldfrag(bK + r16 * 72,        ks * 32 + a0, ks * 32 + a0 + 16);
      v16bf bk1 = ldfrag(bK + (16 + r16) * 72, ks * 32 + a0, ks * 32 + a0 + 16);
      s0 = wmma_bf16f32(qf[ks], bk0, s0);
      s1 = wmma_bf16f32(qf[ks], bk1, s1);
    }

    // ---- online softmax ----
#pragma unroll
    for (int e = 0; e < 8; ++e) {
      const int row_g = qbase + wave * 16 + e + half8;
      float v0 = s0[e] * scale;
      float v1 = s1[e] * scale;
      if (kbase + r16 > row_g)      v0 = -1e30f;
      if (kbase + 16 + r16 > row_g) v1 = -1e30f;
      float mx = fmaxf(v0, v1);
      mx = fmaxf(mx, __shfl_xor(mx, 1, 32));
      mx = fmaxf(mx, __shfl_xor(mx, 2, 32));
      mx = fmaxf(mx, __shfl_xor(mx, 4, 32));
      mx = fmaxf(mx, __shfl_xor(mx, 8, 32));
      const float mnew  = fmaxf(m_i[e], mx);
      const float fcorr = __expf(m_i[e] - mnew);
      m_i[e] = mnew;
      const float p0 = __expf(v0 - mnew);
      const float p1 = __expf(v1 - mnew);
      float sum = p0 + p1;
      sum += __shfl_xor(sum, 1, 32);
      sum += __shfl_xor(sum, 2, 32);
      sum += __shfl_xor(sum, 4, 32);
      sum += __shfl_xor(sum, 8, 32);
      l_i[e] = l_i[e] * fcorr + sum;
      o[0][e] *= fcorr; o[1][e] *= fcorr; o[2][e] *= fcorr; o[3][e] *= fcorr;
      sPw[(e + half8) * 40 + r16]      = (__bf16)p0;
      sPw[(e + half8) * 40 + 16 + r16] = (__bf16)p1;
    }
    asm volatile("s_wait_dscnt 0" ::: "memory");  // per-wave LDS RAW fence for sP

    // ---- O += P V ----
    v16bf pf = ldfrag(sPw + r16 * 40, a0, a0 + 16);
#pragma unroll
    for (int nf = 0; nf < 4; ++nf) {
      v16bf vf = ldfrag(bV + (nf * 16 + r16) * 40, a0, a0 + 16);
      o[nf] = wmma_bf16f32(pf, vf, o[nf]);
    }
  }

  // ---- normalize, store bf16 [B,T,C] for projection GEMM ----
  const int bidx = bh >> 4, h = bh & 15;
#pragma unroll
  for (int e = 0; e < 8; ++e) {
    const float inv = 1.0f / l_i[e];
    const int t = qbase + wave * 16 + e + half8;
    const size_t rowoff = ((size_t)(bidx * SEQ + t)) * N_EMBD + h * HEAD_DIM;
#pragma unroll
    for (int nf = 0; nf < 4; ++nf)
      ob[rowoff + nf * 16 + r16] = (__bf16)(o[nf][e] * inv);
  }
}

/* ---------------- host launch ---------------- */
extern "C" void kernel_launch(void* const* d_in, const int* in_sizes, int n_in,
                              void* d_out, int out_size, void* d_ws, size_t ws_size,
                              hipStream_t stream) {
  const float* x      = (const float*)d_in[0];
  const float* w_attn = (const float*)d_in[1];
  const float* b_attn = (const float*)d_in[2];
  const float* w_proj = (const float*)d_in[3];
  const float* b_proj = (const float*)d_in[4];
  float* out = (float*)d_out;

  char* p = (char*)d_ws;
  __bf16* xb    = (__bf16*)p; p += (size_t)MROWS * N_EMBD * 2;        // 16 MB
  __bf16* wabT  = (__bf16*)p; p += (size_t)N_EMBD * 3 * N_EMBD * 2;   //  6 MB  (transposed)
  __bf16* wpbT  = (__bf16*)p; p += (size_t)N_EMBD * N_EMBD * 2;       //  2 MB  (transposed)
  __bf16* qbuf  = (__bf16*)p; p += (size_t)MROWS * N_EMBD * 2;        // 16 MB [B,H,T,64]
  __bf16* kbuf  = (__bf16*)p; p += (size_t)MROWS * N_EMBD * 2;        // 16 MB [B,H,T,64]
  __bf16* vbuf  = (__bf16*)p; p += (size_t)MROWS * N_EMBD * 2;        // 16 MB [B,H,64,T]
  __bf16* attnb = xb;  // x buffer reused for attention output

  cvt_f32_bf16_kernel<<<2048, 256, 0, stream>>>(x, xb, MROWS * N_EMBD);
  cvt_transpose_kernel<N_EMBD, 3 * N_EMBD><<<2048, 256, 0, stream>>>(w_attn, wabT);
  cvt_transpose_kernel<N_EMBD, N_EMBD><<<1024, 256, 0, stream>>>(w_proj, wpbT);

  gemm_bf16_kernel<0><<<dim3(3 * N_EMBD / 256, MROWS / 128), 256, 0, stream>>>(
      xb, wabT, b_attn, 3 * N_EMBD, qbuf, kbuf, vbuf, nullptr);

  attn_kernel<<<BATCH * N_HEAD * (SEQ / 64), 128, 0, stream>>>(qbuf, kbuf, vbuf, attnb);

  gemm_bf16_kernel<1><<<dim3(N_EMBD / 256, MROWS / 128), 256, 0, stream>>>(
      attnb, wpbT, b_proj, N_EMBD, nullptr, nullptr, nullptr, out);
}
